// Head_75883482185754
// MI455X (gfx1250) — matrix operand order repaired and checked by reading, not compiled, not run on hardware
//
#include <hip/hip_runtime.h>
#include <hip/hip_bf16.h>

#define B_ 8
#define T_ 2048
#define C_ 1024
#define H_ 64

typedef __bf16 bf16;
typedef __attribute__((ext_vector_type(16))) __bf16 v16bf;
typedef __attribute__((ext_vector_type(8)))  __bf16 v8bf;
typedef __attribute__((ext_vector_type(2)))  __bf16 v2bf;
typedef __attribute__((ext_vector_type(8)))  float  v8f;

union V16U { v16bf v; v8bf h[2]; v2bf p[8]; };
union V8U  { v8bf  v; v2bf p[4]; bf16 e[8]; };

__device__ __forceinline__ bf16 f2bf(float f) {
    union { float f; unsigned u; } in; in.f = f;
    unsigned u = in.u;
    u += 0x7fffu + ((u >> 16) & 1u);            // round-to-nearest-even
    union { unsigned short s; bf16 b; } out;
    out.s = (unsigned short)(u >> 16);
    return out.b;
}

// packed f32x2 -> bf16x2 (hardware v_cvt_pk_bf16_f32 when available)
__device__ __forceinline__ v2bf f2bf2(float lo, float hi) {
#if __has_builtin(__builtin_amdgcn_cvt_pk_bf16_f32)
    auto t = __builtin_amdgcn_cvt_pk_bf16_f32(lo, hi);
    static_assert(sizeof(t) == sizeof(v2bf), "cvt_pk_bf16 size mismatch");
    return __builtin_bit_cast(v2bf, t);
#else
    v2bf r; r[0] = f2bf(lo); r[1] = f2bf(hi); return r;
#endif
}

__device__ __forceinline__ v8f wmma_bf16(v16bf a, v16bf b, v8f c) {
    return __builtin_amdgcn_wmma_f32_16x16x32_bf16(false, a, false, b, (short)0, c, false, false);
}

__device__ __forceinline__ float rowmax16(float v) {
    v = fmaxf(v, __shfl_xor(v, 1, 32));
    v = fmaxf(v, __shfl_xor(v, 2, 32));
    v = fmaxf(v, __shfl_xor(v, 4, 32));
    v = fmaxf(v, __shfl_xor(v, 8, 32));
    return v;
}
__device__ __forceinline__ float rowsum16(float v) {
    v += __shfl_xor(v, 1, 32);
    v += __shfl_xor(v, 2, 32);
    v += __shfl_xor(v, 4, 32);
    v += __shfl_xor(v, 8, 32);
    return v;
}

// ---------------------------------------------------------------------------
// Kernel 0: convert+transpose W (fp32 [1024][64]) -> bf16 Wt ([64][1024]) x3
// ---------------------------------------------------------------------------
__global__ void wt_kernel(const float* __restrict__ Wq, const float* __restrict__ Wk,
                          const float* __restrict__ Wv, bf16* __restrict__ Wt) {
    int idx = blockIdx.x * blockDim.x + threadIdx.x;
    if (idx >= 3 * C_ * H_) return;
    int mat = idx / (C_ * H_);
    int rem = idx % (C_ * H_);
    int r = rem / H_;      // input row (C)
    int c = rem % H_;      // input col (H)
    const float* W = (mat == 0) ? Wq : ((mat == 1) ? Wk : Wv);
    Wt[(size_t)mat * H_ * C_ + (size_t)c * C_ + r] = f2bf(W[(size_t)r * H_ + c]);
}

// ---------------------------------------------------------------------------
// Kernel 1: fused Q/K/V projection. One wave computes 16 rows x 64 cols of
// all three outputs; x fragments built once, reused for 12 WMMAs/step.
// All 12 B-fragments are loaded up front each step so the loads clause
// together and overlap the WMMA chain instead of s_wait_loadcnt 0 per WMMA.
// Q,K stored row-major bf16 [B][T][64]; V stored transposed bf16 [B][64][T].
// ---------------------------------------------------------------------------
__global__ void __launch_bounds__(256)
proj_kernel(const float* __restrict__ x, const bf16* __restrict__ Wt,
            bf16* __restrict__ Q, bf16* __restrict__ K, bf16* __restrict__ Vt) {
    const int wave = threadIdx.x >> 5;
    const int lane = threadIdx.x & 31;
    const int rowBlk = blockIdx.x * 8 + wave;          // 16-row block index
    const int b  = (rowBlk * 16) / T_;
    const int t0 = (rowBlk * 16) % T_;
    const int hf = lane >> 4;                          // 0 / 1 (lane half)
    const int nl = lane & 15;
    const int koffA = hf * 16;                         // A-frag K offset
    const int koffB = hf * 16;                         // B-frag K offset
    const float* xrow = x + ((size_t)b * T_ + t0 + nl) * C_;
    const bf16* Wtq = Wt;
    const bf16* Wtk = Wt + (size_t)H_ * C_;
    const bf16* Wtv = Wt + (size_t)2 * H_ * C_;

    v8f cq[4] = {}, ck[4] = {}, cv[4] = {};

    for (int c = 0; c < C_; c += 32) {
        // A fragment: rows = 16 across lane halves, K = c+{koffA..koffA+15}
        const float4 g0 = *(const float4*)(xrow + c + koffA);
        const float4 g1 = *(const float4*)(xrow + c + koffA + 4);
        const float4 g2 = *(const float4*)(xrow + c + koffA + 8);
        const float4 g3 = *(const float4*)(xrow + c + koffA + 12);

        // batch all 12 B-fragment loads (one clause, deep loadcnt pipeline)
        v16bf bq[4], bk[4], bv[4];
        #pragma unroll
        for (int j = 0; j < 4; ++j) {
            const size_t off = (size_t)(16 * j + nl) * C_ + c + koffB;
            bq[j] = *(const v16bf*)(Wtq + off);
            bk[j] = *(const v16bf*)(Wtk + off);
            bv[j] = *(const v16bf*)(Wtv + off);
        }

        V16U a;
        a.p[0] = f2bf2(g0.x, g0.y); a.p[1] = f2bf2(g0.z, g0.w);
        a.p[2] = f2bf2(g1.x, g1.y); a.p[3] = f2bf2(g1.z, g1.w);
        a.p[4] = f2bf2(g2.x, g2.y); a.p[5] = f2bf2(g2.z, g2.w);
        a.p[6] = f2bf2(g3.x, g3.y); a.p[7] = f2bf2(g3.z, g3.w);

        #pragma unroll
        for (int j = 0; j < 4; ++j) {
            cq[j] = wmma_bf16(a.v, bq[j], cq[j]);
            ck[j] = wmma_bf16(a.v, bk[j], ck[j]);
            cv[j] = wmma_bf16(a.v, bv[j], cv[j]);
        }
    }

    #pragma unroll
    for (int j = 0; j < 4; ++j) {
        const int col = 16 * j + nl;
        // Q, K: row-major scalar b16 stores (row stride 64)
        #pragma unroll
        for (int r = 0; r < 8; ++r) {
            const int m = hf * 8 + r;
            const size_t row = (size_t)b * T_ + t0 + m;
            Q[row * H_ + col] = f2bf(cq[j][r]);
            K[row * H_ + col] = f2bf(ck[j][r]);
        }
        // Vt: a lane's 8 accumulator elements are 8 consecutive t positions
        V8U vs;
        vs.p[0] = f2bf2(cv[j][0], cv[j][1]);
        vs.p[1] = f2bf2(cv[j][2], cv[j][3]);
        vs.p[2] = f2bf2(cv[j][4], cv[j][5]);
        vs.p[3] = f2bf2(cv[j][6], cv[j][7]);
        *(v8bf*)&Vt[((size_t)b * H_ + col) * T_ + t0 + hf * 8] = vs.v;
    }
}

// ---------------------------------------------------------------------------
// Kernel 2: flash attention with ALiBi + causal mask. One wave = 16 queries,
// streams 32-key blocks up to the causal limit. Online softmax kept in VGPRs;
// P re-shaped to the A-fragment layout through wave-private LDS. V fragments
// are loaded BEFORE the softmax math so global latency hides behind exp/shfl.
// ---------------------------------------------------------------------------
__global__ void __launch_bounds__(256)
attn_kernel(const bf16* __restrict__ Q, const bf16* __restrict__ K,
            const bf16* __restrict__ Vt, float* __restrict__ out) {
    __shared__ __align__(64) bf16 pTile[8][16][32];    // per-wave P staging (8 KB)

    const int wave = threadIdx.x >> 5;
    const int lane = threadIdx.x & 31;
    const int blk  = blockIdx.x;
    const int b    = blk / (T_ / 128);
    const int qb   = (blk % (T_ / 128)) * 128 + wave * 16;
    const int hf   = lane >> 4;
    const int nl   = lane & 15;
    const int koff = hf * 16;

    const float slope = 0.70710678118654752f;          // 2^(-8*1/16)
    const float scale = 0.125f;                        // 1/sqrt(64)

    // Q A-fragments (heads 0..31 and 32..63), loaded once per wave
    const bf16* qrow = Q + ((size_t)b * T_ + qb + nl) * H_;
    V16U aq0, aq1;
    aq0.h[0] = *(const v8bf*)(qrow + koff);
    aq0.h[1] = *(const v8bf*)(qrow + koff + 8);
    aq1.h[0] = *(const v8bf*)(qrow + 32 + koff);
    aq1.h[1] = *(const v8bf*)(qrow + 32 + koff + 8);

    float rm[8], rs[8];
    v8f o[4] = {};
    #pragma unroll
    for (int r = 0; r < 8; ++r) { rm[r] = -1e30f; rs[r] = 0.0f; }

    const bf16* Kb = K  + (size_t)b * T_ * H_;
    const bf16* Vb = Vt + (size_t)b * H_ * T_;

    for (int kb = 0; kb < qb + 16; kb += 32) {
        // --- S = Q K^T for two 16-key tiles (4 WMMAs) ---
        const bf16* k0 = Kb + (size_t)(kb + nl) * H_ + hf * 16;
        const bf16* k1 = Kb + (size_t)(kb + 16 + nl) * H_ + hf * 16;
        __builtin_prefetch(k0 + 32 * H_, 0, 3);        // next key block
        v16bf b00 = *(const v16bf*)(k0);
        v16bf b01 = *(const v16bf*)(k0 + 32);
        v16bf b10 = *(const v16bf*)(k1);
        v16bf b11 = *(const v16bf*)(k1 + 32);
        v8f s0 = {}, s1 = {};
        s0 = wmma_bf16(aq0.v, b00, s0);
        s0 = wmma_bf16(aq1.v, b01, s0);
        s1 = wmma_bf16(aq0.v, b10, s1);
        s1 = wmma_bf16(aq1.v, b11, s1);

        // --- issue V fragment loads early; they retire during softmax math ---
        v16bf bv[4];
        #pragma unroll
        for (int j = 0; j < 4; ++j) {
            const bf16* vp = Vb + (size_t)(16 * j + nl) * T_ + kb + hf * 16;
            __builtin_prefetch(vp + 32, 0, 3);
            bv[j] = *(const v16bf*)vp;
        }

        // --- bias + mask + online softmax (rows span 16 lanes in one VGPR) ---
        float p0[8], p1[8], f[8];
        #pragma unroll
        for (int r = 0; r < 8; ++r) {
            const int q     = qb + hf * 8 + r;
            const int kidx0 = kb + nl;
            const int kidx1 = kb + 16 + nl;
            float v0 = (s0[r] + (float)(kidx0 - q) * slope) * scale;
            float v1 = (s1[r] + (float)(kidx1 - q) * slope) * scale;
            v0 = (kidx0 <= q) ? v0 : -1e30f;
            v1 = (kidx1 <= q) ? v1 : -1e30f;
            float tmax = rowmax16(fmaxf(v0, v1));
            float rn   = fmaxf(rm[r], tmax);
            f[r]  = __expf(rm[r] - rn);
            rm[r] = rn;
            p0[r] = __expf(v0 - rn);
            p1[r] = __expf(v1 - rn);
            rs[r] = rs[r] * f[r] + rowsum16(p0[r] + p1[r]);
        }
        #pragma unroll
        for (int j = 0; j < 4; ++j)
            #pragma unroll
            for (int r = 0; r < 8; ++r) o[j][r] *= f[r];

        // --- stage P (16x32) to wave-private LDS, reload in A-frag layout ---
        #pragma unroll
        for (int r = 0; r < 8; ++r) {
            const int m = hf * 8 + r;
            pTile[wave][m][nl]      = f2bf(p0[r]);
            pTile[wave][m][nl + 16] = f2bf(p1[r]);
        }
        V16U ap;                                       // same-wave DS ops are in-order
        ap.h[0] = *(const v8bf*)&pTile[wave][nl][koff];
        ap.h[1] = *(const v8bf*)&pTile[wave][nl][koff + 8];

        // --- O += P V (4 WMMAs over head tiles) ---
        #pragma unroll
        for (int j = 0; j < 4; ++j)
            o[j] = wmma_bf16(ap.v, bv[j], o[j]);
    }

    // --- normalize and write fp32 output ---
    float* orow = out + ((size_t)b * T_ + qb) * H_;
    #pragma unroll
    for (int r = 0; r < 8; ++r) {
        const int m = hf * 8 + r;
        const float inv = 1.0f / rs[r];
        #pragma unroll
        for (int j = 0; j < 4; ++j)
            orow[(size_t)m * H_ + 16 * j + nl] = o[j][r] * inv;
    }
}

// ---------------------------------------------------------------------------
extern "C" void kernel_launch(void* const* d_in, const int* in_sizes, int n_in,
                              void* d_out, int out_size, void* d_ws, size_t ws_size,
                              hipStream_t stream) {
    const float* x  = (const float*)d_in[0];
    const float* Wq = (const float*)d_in[1];
    const float* Wk = (const float*)d_in[2];
    const float* Wv = (const float*)d_in[3];
    float* out = (float*)d_out;

    char* ws = (char*)d_ws;
    // workspace layout (all 256B aligned): Wt(384K) | Q(2M) | K(2M) | Vt(2M)
    bf16* Wt = (bf16*)(ws);
    bf16* Q  = (bf16*)(ws + 393216);
    bf16* K  = (bf16*)(ws + 393216 + 2097152);
    bf16* Vt = (bf16*)(ws + 393216 + 2 * 2097152);

    wt_kernel<<<(3 * C_ * H_ + 255) / 256, 256, 0, stream>>>(Wq, Wk, Wv, Wt);
    proj_kernel<<<(B_ * T_) / 128, 256, 0, stream>>>(x, Wt, Q, K, Vt);
    attn_kernel<<<(B_ * T_) / 128, 256, 0, stream>>>(Q, K, Vt, out);
}
